// BiLSTM_CRF_46566035423755
// MI455X (gfx1250) — matrix operand order repaired and checked by reading, not compiled, not run on hardware
//
#include <hip/hip_runtime.h>

// ---------------- types / constants ----------------
typedef __attribute__((ext_vector_type(8)))  float  v8f;
typedef __attribute__((ext_vector_type(8)))  __bf16 v8bf;
typedef __attribute__((ext_vector_type(16))) __bf16 v16bf;

#define GLOBAL_AS __attribute__((address_space(1)))
typedef const __bf16 GLOBAL_AS* gbf_cptr;
typedef const v16bf GLOBAL_AS* gv16_cptr;

#define B_    64
#define L_    256
#define E_    256
#define H_    256
#define HID_  512
#define G4H_  1024
#define C_    20
#define CP_   32          // padded class dim for WMMA-friendly feats
#define START_ 18
#define STOP_  19
#define NEGV  (-10000.0f)
#define HP_   264         // padded LDS row stride for h (kills bank conflicts)

static __device__ __forceinline__ v16bf cat16(v8bf lo, v8bf hi) {
  return __builtin_shufflevector(lo, hi, 0,1,2,3,4,5,6,7,8,9,10,11,12,13,14,15);
}
static __device__ __forceinline__ float sigf(float x) {
  return 1.0f / (1.0f + __expf(-x));
}
static __device__ __forceinline__ float tanh_fast(float x) {
  // tanh(x) = 1 - 2/(exp(2x)+1)
  return 1.0f - 2.0f / (__expf(2.0f * x) + 1.0f);
}

// ---------------- 1. weight transpose + f32->bf16 ----------------
// dst[k*dstLd + n] = (n < srcRows) ? src[n*srcCols + k] : 0   (k < srcCols, n < dstLd)
__global__ __launch_bounds__(256)
void k_transpose_bf16(const float* __restrict__ src, __bf16* __restrict__ dst,
                      int srcRows, int srcCols, int dstLd) {
  int total = srcCols * dstLd;
  for (int i = blockIdx.x * blockDim.x + threadIdx.x; i < total;
       i += gridDim.x * blockDim.x) {
    int k = i / dstLd, n = i % dstLd;
    float v = (n < srcRows) ? src[(size_t)n * srcCols + k] : 0.0f;
    dst[i] = (__bf16)v;
  }
}

// ---------------- 2. embedding gather (time-major) + bf16 ----------------
// X[(t*B + b)*E + e] = embed[sent[b*L + t]*E + e]
__global__ __launch_bounds__(256)
void k_gather_embed(const int* __restrict__ sent, const float* __restrict__ embed,
                    __bf16* __restrict__ X) {
  int total4 = L_ * B_ * (E_ / 4);
  for (int i = blockIdx.x * blockDim.x + threadIdx.x; i < total4;
       i += gridDim.x * blockDim.x) {
    int t   = i / (B_ * (E_ / 4));
    int rem = i % (B_ * (E_ / 4));
    int b   = rem / (E_ / 4);
    int e4  = rem % (E_ / 4);
    int tok = sent[b * L_ + t];
    const float4 f = *(const float4*)(embed + (size_t)tok * E_ + e4 * 4);
    __bf16* dst = X + (size_t)(t * B_ + b) * E_ + e4 * 4;
    dst[0] = (__bf16)f.x; dst[1] = (__bf16)f.y;
    dst[2] = (__bf16)f.z; dst[3] = (__bf16)f.w;
  }
}

// ---------------- 3. bf16 WMMA GEMM: C = A * Bt + bias ----------------
// A: M x K row-major bf16.  Bt: K x N row-major bf16 (original W transposed).
// Each wave computes a (16*MT) x (16*NT) macro-tile, K-stepped by 32, f32 acc.
// MT/NT are compile-time so every WMMA is unconditional (no acc phi copies).
template <int MT, int NT>
__global__ __launch_bounds__(256)
void k_wmma_gemm(const __bf16* __restrict__ A, const __bf16* __restrict__ Bt,
                 const float* __restrict__ bias, int biasN,
                 float* __restrict__ Cm, int M, int N, int K) {
  int wave = threadIdx.x >> 5, lane = threadIdx.x & 31;
  int lh = lane & 15, hl = lane >> 4;
  int nU = N / (16 * NT);
  int u = blockIdx.x * 8 + wave;
  if (u >= (M / (16 * MT)) * nU) return;
  int rt = u / nU, cu = u % nU;
  int m0 = rt * 16 * MT, n0 = cu * 16 * NT;

  v8f acc[MT][NT];
  #pragma unroll
  for (int i = 0; i < MT; i++)
    #pragma unroll
    for (int j = 0; j < NT; j++)
      #pragma unroll
      for (int v = 0; v < 8; v++) acc[i][j][v] = 0.0f;

  const __bf16* arow = A + (size_t)(m0 + lh) * K + 8 * hl;
  for (int k0 = 0; k0 < K; k0 += 32) {
    v16bf a[MT];
    #pragma unroll
    for (int i = 0; i < MT; i++) {
      const __bf16* ap = arow + (size_t)(i * 16) * K + k0;
      a[i] = cat16(*(const v8bf*)(ap), *(const v8bf*)(ap + 16));
    }
    const __bf16* brow = Bt + (size_t)(k0 + 8 * hl) * N + n0 + lh;
    #pragma unroll
    for (int j = 0; j < NT; j++) {
      v16bf b;
      #pragma unroll
      for (int s = 0; s < 16; s++)
        b[s] = brow[(size_t)(s < 8 ? s : s + 8) * N + j * 16];
      #pragma unroll
      for (int i = 0; i < MT; i++)
        acc[i][j] = __builtin_amdgcn_wmma_f32_16x16x32_bf16(
            false, a[i], false, b, (short)0, acc[i][j], false, false);
    }
  }
  #pragma unroll
  for (int i = 0; i < MT; i++)
    #pragma unroll
    for (int j = 0; j < NT; j++) {
      int n = n0 + j * 16 + lh;
      float bv = (n < biasN) ? bias[n] : 0.0f;
      #pragma unroll
      for (int v = 0; v < 8; v++) {
        int m = m0 + i * 16 + v + 8 * hl;
        Cm[(size_t)m * N + n] = acc[i][j][v] + bv;
      }
    }
}

// ---------------- 4. persistent per-direction LSTM recurrence ----------------
// One workgroup (32 waves) per direction. h lives in LDS (bf16, padded),
// c lives in registers. Each wave owns rows rt*16..+15 and h-columns
// jt*16..jt*16+31 ACROSS ALL FOUR GATES, so the cell math is register-only.
// All weight/xp/hs accesses are single-base + compile-time immediate offsets
// (IOFFSET is 24-bit signed, max used here is ~507KB) so the hot loop carries
// no 64-bit address arithmetic. The weight base is laundered through empty
// inline-asm each step (blocks LICM hoist -> scratch spill) and typed
// address_space(1) so loads lower to per-element 16-bit global loads (LOADcnt only).
__global__ __launch_bounds__(1024)
void k_lstm_rec(const float* __restrict__ xp_f, const float* __restrict__ xp_b,
                const __bf16* __restrict__ WhhT_f, const __bf16* __restrict__ WhhT_b,
                const float* __restrict__ h0, const float* __restrict__ c0,
                __bf16* __restrict__ hs) {
  const int dir = blockIdx.x;
  const float*  xp = dir ? xp_b   : xp_f;
  const __bf16* Wt = dir ? WhhT_b : WhhT_f;
  extern __shared__ __align__(16) char smem_raw[];
  __bf16* hsh = (__bf16*)smem_raw;                 // [B_][HP_]

  int tid = threadIdx.x;
  int wave = tid >> 5, lane = tid & 31, lh = lane & 15, hl = lane >> 4;
  int rt = wave & 3;            // row tile (M = 64 -> 4 tiles)
  int jt = (wave >> 2) * 2;     // base j-tile (h cols), 2 tiles per wave
  int mrow = rt * 16 + 8 * hl;  // first M row this half-lane owns (v adds 0..7)
  int jcol = jt * 16 + lh;      // first h column this lane owns (q adds 0/16)

  const float* h0d = h0 + (size_t)dir * B_ * H_;
  const float* c0d = c0 + (size_t)dir * B_ * H_;
  for (int i = tid; i < B_ * H_; i += 1024) {
    int m = i >> 8, j = i & 255;
    hsh[m * HP_ + j] = (__bf16)h0d[i];
  }
  v8f creg[2];
  #pragma unroll
  for (int q = 0; q < 2; q++)
    #pragma unroll
    for (int v = 0; v < 8; v++)
      creg[q][v] = c0d[(mrow + v) * H_ + jcol + q * 16];
  __syncthreads();

  // Per-wave bases folded once; hot-loop accesses use immediate offsets only.
  gbf_cptr Wbase0 = (gbf_cptr)Wt + (size_t)(8 * hl) * G4H_ + jt * 16 + lh;
  const __bf16* ards = hsh + (rt * 16 + lh) * HP_ + 8 * hl;
  __bf16* hlds = hsh + (size_t)mrow * HP_ + jcol;

  for (int s = 0; s < L_; s++) {
    int t = dir ? (L_ - 1 - s) : s;
    if (s + 1 < L_) {   // warm L2 for next step's gate pre-activations
      int tn = dir ? (L_ - 2 - s) : (s + 1);
      __builtin_prefetch((const void*)(xp + (size_t)tn * B_ * G4H_ + tid * 16), 0, 1);
    }

    // Opaque copy of the folded weight base: blocks LICM of the
    // time-invariant fragment loads without adding per-step address math.
    gbf_cptr Wb = Wbase0;
    asm volatile("" : "+v"(Wb));

    v8f acc[4][2];
    #pragma unroll
    for (int g = 0; g < 4; g++)
      #pragma unroll
      for (int q = 0; q < 2; q++)
        #pragma unroll
        for (int v = 0; v < 8; v++) acc[g][q][v] = 0.0f;

    #pragma unroll
    for (int k0 = 0; k0 < H_; k0 += 32) {
      v8bf alo = *(const v8bf*)(ards + k0);        // ds_load_b128
      v8bf ahi = *(const v8bf*)(ards + k0 + 16);
      v16bf a = cat16(alo, ahi);
      #pragma unroll
      for (int g = 0; g < 4; g++)
        #pragma unroll
        for (int q = 0; q < 2; q++) {
          // element offset: (k0+kk(s))*1024 + g*256 + q*16 (constant after unroll)
          v16bf b;
          #pragma unroll
          for (int s = 0; s < 16; s++)
            b[s] = Wb[(k0 + (s < 8 ? s : s + 8)) * G4H_ + g * 256 + q * 16];
          acc[g][q] = __builtin_amdgcn_wmma_f32_16x16x32_bf16(
              false, a, false, b, (short)0, acc[g][q], false, false);
        }
    }

    // gate pre-activations: one base, constant offsets v*1024 + g*256 + q*16
    const float* xrb = xp + (size_t)t * B_ * G4H_ + (size_t)mrow * G4H_ + jcol;
    float hnew[2][8];
    #pragma unroll
    for (int q = 0; q < 2; q++)
      #pragma unroll
      for (int v = 0; v < 8; v++) {
        float gi = acc[0][q][v] + xrb[v * G4H_ + q * 16];
        float gf = acc[1][q][v] + xrb[v * G4H_ + q * 16 + H_];
        float gg = acc[2][q][v] + xrb[v * G4H_ + q * 16 + 2 * H_];
        float go = acc[3][q][v] + xrb[v * G4H_ + q * 16 + 3 * H_];
        float cc = sigf(gf) * creg[q][v] + sigf(gi) * tanh_fast(gg);
        creg[q][v] = cc;
        hnew[q][v] = sigf(go) * tanh_fast(cc);
      }

    __syncthreads();   // all waves done reading old h
    __bf16* hgb = hs + ((size_t)t * B_ + mrow) * HID_ + dir * H_ + jcol;
    #pragma unroll
    for (int q = 0; q < 2; q++)
      #pragma unroll
      for (int v = 0; v < 8; v++) {
        __bf16 hb = (__bf16)hnew[q][v];
        hlds[v * HP_ + q * 16] = hb;         // ds_store_b16, imm offsets
        hgb[v * HID_ + q * 16] = hb;         // global_store_b16, imm offsets
      }
    __syncthreads();   // new h visible before next step
  }
}

// ---------------- 5. CRF forward scan ----------------
__global__ __launch_bounds__(256)
void k_crf_forward(const float* __restrict__ feats, const float* __restrict__ trans,
                   float* __restrict__ fwd) {
  __shared__ float alpha[2][B_ * C_];
  __shared__ float tsh[C_ * C_];
  int tid = threadIdx.x;
  for (int i = tid; i < C_ * C_; i += 256) tsh[i] = trans[i];
  for (int i = tid; i < B_ * C_; i += 256)
    alpha[0][i] = ((i % C_) == START_) ? 0.0f : NEGV;
  __syncthreads();
  int cur = 0;
  for (int t = 0; t < L_; t++) {
    const float* ft = feats + (size_t)t * B_ * CP_;
    for (int i = tid; i < B_ * C_; i += 256) {
      int b = i / C_, cto = i % C_;
      const float* ar = &alpha[cur][b * C_];
      const float* tr = &tsh[cto * C_];
      float mx = -3.0e38f;
      #pragma unroll
      for (int cf = 0; cf < C_; cf++) mx = fmaxf(mx, ar[cf] + tr[cf]);
      float sum = 0.0f;
      #pragma unroll
      for (int cf = 0; cf < C_; cf++) sum += __expf(ar[cf] + tr[cf] - mx);
      alpha[cur ^ 1][i] = mx + __logf(sum) + ft[b * CP_ + cto];
    }
    __syncthreads();
    cur ^= 1;
  }
  if (tid < B_) {
    const float* ar = &alpha[cur][tid * C_];
    const float* tr = &tsh[STOP_ * C_];
    float mx = -3.0e38f;
    #pragma unroll
    for (int c = 0; c < C_; c++) mx = fmaxf(mx, ar[c] + tr[c]);
    float sum = 0.0f;
    #pragma unroll
    for (int c = 0; c < C_; c++) sum += __expf(ar[c] + tr[c] - mx);
    fwd[tid] = mx + __logf(sum);
  }
}

// ---------------- 6. gold path score ----------------
__global__ __launch_bounds__(64)
void k_gold(const int* __restrict__ tags, const float* __restrict__ feats,
            const float* __restrict__ trans, float* __restrict__ gold) {
  int b = threadIdx.x;
  if (b >= B_) return;
  int prev = START_;
  float s = 0.0f;
  for (int t = 0; t < L_; t++) {
    int tg = tags[b * L_ + t];
    s += trans[tg * C_ + prev] + feats[((size_t)t * B_ + b) * CP_ + tg];
    prev = tg;
  }
  s += trans[STOP_ * C_ + prev];
  gold[b] = s;
}

// ---------------- 7. final mean reduce ----------------
__global__ __launch_bounds__(64)
void k_final(const float* __restrict__ fwd, const float* __restrict__ gold,
             float* __restrict__ out) {
  __shared__ float red[B_];
  int tid = threadIdx.x;
  red[tid] = fwd[tid] - gold[tid];
  __syncthreads();
  for (int off = B_ / 2; off > 0; off >>= 1) {
    if (tid < off) red[tid] += red[tid + off];
    __syncthreads();
  }
  if (tid == 0) out[0] = red[0] / (float)B_;
}

// ---------------- launch ----------------
extern "C" void kernel_launch(void* const* d_in, const int* in_sizes, int n_in,
                              void* d_out, int out_size, void* d_ws, size_t ws_size,
                              hipStream_t stream) {
  const int*   sent   = (const int*)  d_in[0];
  const int*   tags   = (const int*)  d_in[1];
  const float* embed  = (const float*)d_in[2];
  const float* Wih_f  = (const float*)d_in[3];
  const float* Whh_f  = (const float*)d_in[4];
  const float* b_f    = (const float*)d_in[5];
  const float* Wih_b  = (const float*)d_in[6];
  const float* Whh_b  = (const float*)d_in[7];
  const float* b_b    = (const float*)d_in[8];
  const float* W_out  = (const float*)d_in[9];
  const float* b_out  = (const float*)d_in[10];
  const float* trans  = (const float*)d_in[11];
  const float* h0     = (const float*)d_in[12];
  const float* c0     = (const float*)d_in[13];
  float* out = (float*)d_out;

  size_t off = 0;
  auto walloc = [&](size_t bytes) -> char* {
    char* p = (char*)d_ws + off;
    off = (off + bytes + 255) & ~(size_t)255;
    return p;
  };
  __bf16* WihTf = (__bf16*)walloc((size_t)E_  * G4H_ * 2);
  __bf16* WihTb = (__bf16*)walloc((size_t)E_  * G4H_ * 2);
  __bf16* WhhTf = (__bf16*)walloc((size_t)H_  * G4H_ * 2);
  __bf16* WhhTb = (__bf16*)walloc((size_t)H_  * G4H_ * 2);
  __bf16* WoutT = (__bf16*)walloc((size_t)HID_ * CP_ * 2);
  __bf16* Xbf   = (__bf16*)walloc((size_t)L_ * B_ * E_ * 2);
  float*  XPf   = (float*) walloc((size_t)L_ * B_ * G4H_ * 4);
  float*  XPb   = (float*) walloc((size_t)L_ * B_ * G4H_ * 4);
  __bf16* HSbf  = (__bf16*)walloc((size_t)L_ * B_ * HID_ * 2);
  float*  FEATS = (float*) walloc((size_t)L_ * B_ * CP_ * 4);
  float*  FWD   = (float*) walloc(B_ * 4);
  float*  GOLD  = (float*) walloc(B_ * 4);

  // 1. weight prep (f32 -> bf16, K-major so WMMA B-fragments are contiguous)
  k_transpose_bf16<<<dim3(512), dim3(256), 0, stream>>>(Wih_f, WihTf, G4H_, E_, G4H_);
  k_transpose_bf16<<<dim3(512), dim3(256), 0, stream>>>(Wih_b, WihTb, G4H_, E_, G4H_);
  k_transpose_bf16<<<dim3(512), dim3(256), 0, stream>>>(Whh_f, WhhTf, G4H_, H_, G4H_);
  k_transpose_bf16<<<dim3(512), dim3(256), 0, stream>>>(Whh_b, WhhTb, G4H_, H_, G4H_);
  k_transpose_bf16<<<dim3(64),  dim3(256), 0, stream>>>(W_out, WoutT, C_, HID_, CP_);

  // 2. embedding gather (time-major, bf16)
  k_gather_embed<<<dim3(2048), dim3(256), 0, stream>>>(sent, embed, Xbf);

  // 3. input projections: xp = X @ WihT + b  (M=16384, N=1024, K=256)
  //    32x64 macro-tile per wave -> (16384/32)*(1024/64) = 8192 units / 8 waves
  k_wmma_gemm<2, 4><<<dim3(1024), dim3(256), 0, stream>>>(
      Xbf, WihTf, b_f, G4H_, XPf, L_ * B_, G4H_, E_);
  k_wmma_gemm<2, 4><<<dim3(1024), dim3(256), 0, stream>>>(
      Xbf, WihTb, b_b, G4H_, XPb, L_ * B_, G4H_, E_);

  // 4. bidirectional recurrence: 1 persistent workgroup per direction
  k_lstm_rec<<<dim3(2), dim3(1024), B_ * HP_ * 2, stream>>>(
      XPf, XPb, WhhTf, WhhTb, h0, c0, HSbf);

  // 5. output projection: feats = HS @ WoutT + b_out  (N padded to 32)
  //    32x32 macro-tile per wave -> (16384/32)*(32/32) = 512 units / 8 waves
  k_wmma_gemm<2, 2><<<dim3(64), dim3(256), 0, stream>>>(
      HSbf, WoutT, b_out, C_, FEATS, L_ * B_, CP_, HID_);

  // 6-8. CRF forward scan, gold score, mean
  k_crf_forward<<<dim3(1), dim3(256), 0, stream>>>(FEATS, trans, FWD);
  k_gold<<<dim3(1), dim3(64), 0, stream>>>(tags, FEATS, trans, GOLD);
  k_final<<<dim3(1), dim3(64), 0, stream>>>(FWD, GOLD, out);
}
